// RWKVv7Separator_34333968564676
// MI455X (gfx1250) — matrix operand order repaired and checked by reading, compiled, not run on hardware
//
#include <hip/hip_runtime.h>
#include <hip/hip_bf16.h>
#include <math.h>

#define BQ 2
#define TQ 1024
#define CIN 512
#define HD 256
#define HNH 4
#define HEADSZ 64
#define NTOK (BQ*TQ)

typedef __attribute__((ext_vector_type(16))) __bf16 v16bf;
typedef __attribute__((ext_vector_type(8)))  float  v8f;

enum { EPI_NONE=0, EPI_BIAS=1, EPI_TANH=2, EPI_SIGMOID=3, EPI_BIAS_SIGMOID=4,
       EPI_RELU_SQ=5, EPI_ADD=6 };

// ---------------------------------------------------------------------------
// Weight pre-pack: W[K,N] f32 -> bf16 fragments in exact WMMA B-operand order.
// Fragment (ntile, kb32, lane) holds 16 bf16: W[kb32*32 + bhalf + e + (e>=8?8:0)][ntile*16 + l15]
// with bhalf = (lane>=16)?8:0 (B layout matching A: lanes 0-15 K={0..7,16..23},
// lanes 16-31 K={8..15,24..31}, two K per VGPR). Linear index = (ntile*K32+kb32)*32+lane.
// ---------------------------------------------------------------------------
__global__ void packW_k(const float* __restrict__ W, __bf16* __restrict__ Wp,
                        int K, int N) {
  const int idx = blockIdx.x * blockDim.x + threadIdx.x;
  const int K32 = K >> 5;
  const int total = (N >> 4) * K32 * 32;
  if (idx >= total) return;
  const int lane  = idx & 31;
  const int rest  = idx >> 5;
  const int kb32  = rest % K32;
  const int ntile = rest / K32;
  const int bhalf = (lane >= 16) ? 8 : 0;
  const int col   = ntile * 16 + (lane & 15);
  const float* src = W + (size_t)(kb32 * 32 + bhalf) * N + col;
  __bf16* dst = Wp + (size_t)idx * 16;
#pragma unroll
  for (int e = 0; e < 16; ++e) dst[e] = (__bf16)src[(size_t)(e + (e >= 8 ? 8 : 0)) * N];
}

// ---------------------------------------------------------------------------
// GEMM: Out[M,N](f32) = epi(A[M,K](f32) @ W[K,N] [+bias][+Add])
// One wave computes a 16 x (16*NTW) strip: NTW WMMAs per K-step share one A
// fragment; B fragments come pre-packed in bf16 (two b128 loads, no cvt).
// blockDim=256 (8 waves); grid = tiles/8 exactly (M=2048 -> Mtiles=128), so
// EXEC is all-ones for every WMMA.
//   A (16x32 bf16): lane<16 -> M=lane, K={0..7,16..23}+kb;
//                   lane>=16 -> M=lane-16, K={8..15,24..31}+kb
//   C/D (16x16 f32): lane holds col N=lane&15; VGPR r -> M=r (+8 if lane>=16)
// ---------------------------------------------------------------------------
template<int EPI, int NTW>
__global__ void gemm_wmma(const float* __restrict__ A, const __bf16* __restrict__ Wp,
                          const float* __restrict__ bias, const float* __restrict__ Add,
                          float* __restrict__ Out, int M, int N, int K) {
  const int wave   = threadIdx.x >> 5;
  const int lane   = threadIdx.x & 31;
  const int tilesN = N / (16 * NTW);
  const int tile   = blockIdx.x * 8 + wave;
  const int tm     = (tile / tilesN) << 4;
  const int tnb    = (tile % tilesN) * (16 * NTW);
  const int l15    = lane & 15;
  const int ahalf  = (lane >> 4) << 3;
  const int K32    = K >> 5;
  const int ntile0 = tnb >> 4;

  const v8f zero = {0.f,0.f,0.f,0.f,0.f,0.f,0.f,0.f};
  v8f acc[NTW];
#pragma unroll
  for (int j = 0; j < NTW; ++j) acc[j] = zero;

  const float* arow = A + (size_t)(tm + l15) * K + ahalf;

  for (int kb32 = 0; kb32 < K32; ++kb32) {
    union { v16bf v; uint4 q[2]; } a;
    const float* ap = arow + (kb32 << 5);
    __builtin_prefetch(ap + 32, 0, 0);           // global_prefetch_b8 next K-slab
#pragma unroll
    for (int e = 0; e < 8; ++e) a.v[e]     = (__bf16)ap[e];
#pragma unroll
    for (int e = 0; e < 8; ++e) a.v[8 + e] = (__bf16)ap[16 + e];
#pragma unroll
    for (int j = 0; j < NTW; ++j) {
      union { v16bf v; uint4 q[2]; } b;
      const __bf16* bp = Wp + ((((size_t)(ntile0 + j) * K32 + kb32) << 5) + lane) * 16;
      b.q[0] = *(const uint4*)bp;
      b.q[1] = *(const uint4*)(bp + 8);
      acc[j] = __builtin_amdgcn_wmma_f32_16x16x32_bf16(false, a.v, false, b.v,
                                                       (short)0, acc[j], false, false);
    }
  }

  const int mbase = tm + ahalf;
#pragma unroll
  for (int j = 0; j < NTW; ++j) {
    const int col = tnb + (j << 4) + l15;
    float bv = 0.f;
    if (EPI == EPI_BIAS || EPI == EPI_BIAS_SIGMOID) bv = bias[col];
#pragma unroll
    for (int r = 0; r < 8; ++r) {
      const int row = mbase + r;
      float z = acc[j][r];
      if (EPI == EPI_BIAS || EPI == EPI_BIAS_SIGMOID) z += bv;
      if (EPI == EPI_TANH) z = tanhf(z);
      if (EPI == EPI_SIGMOID || EPI == EPI_BIAS_SIGMOID) z = 1.f / (1.f + expf(-z));
      if (EPI == EPI_RELU_SQ) { z = z > 0.f ? z : 0.f; z = z * z; }
      if (EPI == EPI_ADD) z += Add[(size_t)row * N + col];
      Out[(size_t)row * N + col] = z;
    }
  }
}

// ---------------- LayerNorm: one wave per token ----------------------------
__global__ void layernorm_k(const float* __restrict__ x, const float* __restrict__ w,
                            const float* __restrict__ b, float* __restrict__ out, int C) {
  const int n = blockIdx.x;
  const int lane = threadIdx.x;
  const float* xr = x + (size_t)n * C;
  float s = 0.f;
  for (int c = lane; c < C; c += 32) s += xr[c];
  for (int o = 16; o > 0; o >>= 1) s += __shfl_xor(s, o, 32);
  const float mu = s / (float)C;
  float v = 0.f;
  for (int c = lane; c < C; c += 32) { float d = xr[c] - mu; v += d * d; }
  for (int o = 16; o > 0; o >>= 1) v += __shfl_xor(v, o, 32);
  const float rstd = rsqrtf(v / (float)C + 1e-5f);
  float* orow = out + (size_t)n * C;
  for (int c = lane; c < C; c += 32) orow[c] = (xr[c] - mu) * rstd * w[c] + b[c];
}

// ---------------- time-shift + 6-way mix (tmix front end) -------------------
__global__ void mix6_k(const float* __restrict__ x,
                       const float* cr, const float* cw, const float* ck,
                       const float* cv, const float* ca, const float* cg,
                       float* o0, float* o1, float* o2,
                       float* o3, float* o4, float* o5) {
  const int n = blockIdx.x, c = threadIdx.x;
  const int t = n % TQ;
  const size_t idx = (size_t)n * HD + c;
  const float xv = x[idx];
  const float prev = (t > 0) ? x[idx - HD] : 0.f;
  const float xx = prev - xv;
  o0[idx] = xv + xx * cr[c];
  o1[idx] = xv + xx * cw[c];
  o2[idx] = xv + xx * ck[c];
  o3[idx] = xv + xx * cv[c];
  o4[idx] = xv + xx * ca[c];
  o5[idx] = xv + xx * cg[c];
}

__global__ void mix1_k(const float* __restrict__ x, const float* ck, float* o) {
  const int n = blockIdx.x, c = threadIdx.x;
  const int t = n % TQ;
  const size_t idx = (size_t)n * HD + c;
  const float xv = x[idx];
  const float prev = (t > 0) ? x[idx - HD] : 0.f;
  o[idx] = xv + (prev - xv) * ck[c];
}

// ---------------- elementwise helpers --------------------------------------
__global__ void ew_wdecay(float* z, int n) {   // z = -softplus(-z) - 0.5
  int i = blockIdx.x * blockDim.x + threadIdx.x;
  if (i >= n) return;
  const float x = z[i];
  const float sp = fmaxf(-x, 0.f) + log1pf(expf(-fabsf(x)));
  z[i] = -sp - 0.5f;
}
__global__ void ew_vmix(float* v, const float* vf, const float* vs, int n) {
  int i = blockIdx.x * blockDim.x + threadIdx.x;
  if (i >= n) return;
  v[i] = v[i] + (vf[i] - v[i]) * vs[i];
}
__global__ void ew_copy(const float* a, float* o, int n) {
  int i = blockIdx.x * blockDim.x + threadIdx.x;
  if (i < n) o[i] = a[i];
}
__global__ void ew_avg(const float* a, const float* b, float* o, int n) {
  int i = blockIdx.x * blockDim.x + threadIdx.x;
  if (i < n) o[i] = 0.5f * (a[i] + b[i]);
}
__global__ void flipT_k(const float* __restrict__ in, float* __restrict__ out, int C) {
  int i = blockIdx.x * blockDim.x + threadIdx.x;
  const int total = NTOK * C;
  if (i >= total) return;
  const int c = i % C;
  const int n = i / C;
  const int t = n % TQ, b = n / TQ;
  out[((size_t)(b * TQ + (TQ - 1 - t))) * C + c] = in[i];
}
__global__ void snake_k(float* u, const float* alpha_p, int n) {
  int i = blockIdx.x * blockDim.x + threadIdx.x;
  if (i >= n) return;
  const float a = alpha_p[0] + 1e-9f;
  const float s = sinf(a * u[i]);
  u[i] = u[i] + s * s / a;
}

// ---------------- kk normalize + k update (per token,head; one wave) --------
__global__ void kk_k(const float* __restrict__ k, const float* __restrict__ a,
                     const float* __restrict__ k_k, const float* __restrict__ k_a,
                     float* __restrict__ kk, float* __restrict__ kout) {
  const int nh = blockIdx.x;
  const int h = nh & (HNH - 1);
  const int n = nh >> 2;
  const int lane = threadIdx.x;
  const size_t base = (size_t)n * HD + h * HEADSZ;
  const int hc = h * HEADSZ;
  const float k0 = k[base + lane], k1 = k[base + lane + 32];
  const float x0 = k0 * k_k[hc + lane], x1 = k1 * k_k[hc + lane + 32];
  float ss = x0 * x0 + x1 * x1;
  for (int o = 16; o > 0; o >>= 1) ss += __shfl_xor(ss, o, 32);
  const float inv = 1.f / fmaxf(sqrtf(ss), 1e-12f);
  kk[base + lane] = x0 * inv;
  kk[base + lane + 32] = x1 * inv;
  const float a0 = a[base + lane], a1 = a[base + lane + 32];
  kout[base + lane]      = k0 * (1.f + (a0 - 1.f) * k_a[hc + lane]);
  kout[base + lane + 32] = k1 * (1.f + (a1 - 1.f) * k_a[hc + lane + 32]);
}

// ---------------- WKV7 recurrence: block=64, one block per (b,h) ------------
__global__ void wkv7_k(const float* __restrict__ r, const float* __restrict__ w,
                       const float* __restrict__ k, const float* __restrict__ v,
                       const float* __restrict__ kkv, const float* __restrict__ a,
                       float* __restrict__ out) {
  const int bh = blockIdx.x;
  const int b = bh / HNH, h = bh % HNH;
  const int i = threadIdx.x;
  __shared__ float sr[HEADSZ], sw[HEADSZ], sk[HEADSZ], sv[HEADSZ],
                   skk[HEADSZ], sa[HEADSZ];
  float S[HEADSZ];
#pragma unroll
  for (int j = 0; j < HEADSZ; ++j) S[j] = 0.f;
  for (int t = 0; t < TQ; ++t) {
    const size_t base = ((size_t)(b * TQ + t)) * HD + h * HEADSZ;
    sr[i]  = r[base + i];
    sw[i]  = expf(-expf(w[base + i]));
    sk[i]  = k[base + i];
    sv[i]  = v[base + i];
    skk[i] = kkv[base + i];
    sa[i]  = a[base + i];
    __syncthreads();
    float sad = 0.f;
#pragma unroll
    for (int j = 0; j < HEADSZ; ++j) sad += S[j] * (-skk[j]);   // S @ a, a = -kk
    const float vi = sv[i];
    float y = 0.f;
#pragma unroll
    for (int j = 0; j < HEADSZ; ++j) {
      const float bj = skk[j] * sa[j];                          // b = kk * a_sig
      const float Sn = S[j] * sw[j] + sad * bj + vi * sk[j];
      S[j] = Sn;
      y += Sn * sr[j];
    }
    out[base + i] = y;
    __syncthreads();
  }
}

// ------- per-head GroupNorm + r_k bonus + gate (one wave per token,head) ----
__global__ void post_k(const float* __restrict__ wkv, const float* __restrict__ lnw,
                       const float* __restrict__ lnb, const float* __restrict__ r,
                       const float* __restrict__ k, const float* __restrict__ v,
                       const float* __restrict__ g, const float* __restrict__ rk,
                       float* __restrict__ out) {
  const int nh = blockIdx.x;
  const int h = nh & (HNH - 1);
  const int n = nh >> 2;
  const int lane = threadIdx.x;
  const size_t base = (size_t)n * HD + h * HEADSZ;
  const int hc = h * HEADSZ;
  const float x0 = wkv[base + lane], x1 = wkv[base + lane + 32];
  float s = x0 + x1;
  for (int o = 16; o > 0; o >>= 1) s += __shfl_xor(s, o, 32);
  const float mu = s * (1.f / 64.f);
  float vv = (x0 - mu) * (x0 - mu) + (x1 - mu) * (x1 - mu);
  for (int o = 16; o > 0; o >>= 1) vv += __shfl_xor(vv, o, 32);
  const float rstd = rsqrtf(vv * (1.f / 64.f) + 64e-5f);
  float dot = r[base + lane] * k[base + lane] * rk[hc + lane]
            + r[base + lane + 32] * k[base + lane + 32] * rk[hc + lane + 32];
  for (int o = 16; o > 0; o >>= 1) dot += __shfl_xor(dot, o, 32);
  const float n0 = (x0 - mu) * rstd * lnw[hc + lane] + lnb[hc + lane];
  const float n1 = (x1 - mu) * rstd * lnw[hc + lane + 32] + lnb[hc + lane + 32];
  out[base + lane]      = (n0 + dot * v[base + lane])      * g[base + lane];
  out[base + lane + 32] = (n1 + dot * v[base + lane + 32]) * g[base + lane + 32];
}

// ---------------- mask softmax + final combine ------------------------------
__global__ void final_k(const float* __restrict__ z, const float* __restrict__ mm,
                        const float* __restrict__ r1, const float* __restrict__ r2,
                        float* __restrict__ out) {
  const int n = blockIdx.x;
  const size_t NTC = (size_t)NTOK * CIN;
  for (int c = threadIdx.x; c < CIN; c += blockDim.x) {
    const float l0 = mm[(size_t)n * (2 * CIN) + 2 * c];
    const float l1 = mm[(size_t)n * (2 * CIN) + 2 * c + 1];
    const float mx = fmaxf(l0, l1);
    const float e0 = expf(l0 - mx), e1 = expf(l1 - mx);
    const float inv = 1.f / (e0 + e1);
    const float m1 = e0 * inv, m2 = e1 * inv;
    const size_t idx = (size_t)n * CIN + c;
    const float zv = z[idx], r1v = r1[idx], r2v = r2[idx];
    out[0 * NTC + idx] = m1 * zv + r1v;
    out[1 * NTC + idx] = m2 * zv + r2v;
    out[2 * NTC + idx] = m1;
    out[3 * NTC + idx] = m2;
    out[4 * NTC + idx] = r1v;
    out[5 * NTC + idx] = r2v;
  }
}

// ===========================================================================
// Host-side orchestration
// ===========================================================================
template<int NTW>
static inline void launch_gemm_ntw(hipStream_t s, int epi, const float* A,
                                   const __bf16* Wp, const float* bias,
                                   const float* add, float* out, int M, int N, int K) {
  const int tiles = (M / 16) * (N / (16 * NTW));
  dim3 g(tiles / 8), b(256);
  switch (epi) {
    case EPI_NONE:         gemm_wmma<EPI_NONE, NTW>        <<<g, b, 0, s>>>(A, Wp, bias, add, out, M, N, K); break;
    case EPI_BIAS:         gemm_wmma<EPI_BIAS, NTW>        <<<g, b, 0, s>>>(A, Wp, bias, add, out, M, N, K); break;
    case EPI_TANH:         gemm_wmma<EPI_TANH, NTW>        <<<g, b, 0, s>>>(A, Wp, bias, add, out, M, N, K); break;
    case EPI_SIGMOID:      gemm_wmma<EPI_SIGMOID, NTW>     <<<g, b, 0, s>>>(A, Wp, bias, add, out, M, N, K); break;
    case EPI_BIAS_SIGMOID: gemm_wmma<EPI_BIAS_SIGMOID, NTW><<<g, b, 0, s>>>(A, Wp, bias, add, out, M, N, K); break;
    case EPI_RELU_SQ:      gemm_wmma<EPI_RELU_SQ, NTW>     <<<g, b, 0, s>>>(A, Wp, bias, add, out, M, N, K); break;
    case EPI_ADD:          gemm_wmma<EPI_ADD, NTW>         <<<g, b, 0, s>>>(A, Wp, bias, add, out, M, N, K); break;
  }
}

static __bf16* g_wpack = nullptr;   // set once per kernel_launch call (scratch)

static inline void launch_gemm(hipStream_t s, int epi, const float* A, const float* W,
                               const float* bias, const float* add, float* out,
                               int M, int N, int K) {
  const int frag = (N / 16) * (K / 32) * 32;
  packW_k<<<(frag + 255) / 256, 256, 0, s>>>(W, g_wpack, K, N);
  if (N % 64 == 0) launch_gemm_ntw<4>(s, epi, A, g_wpack, bias, add, out, M, N, K);
  else             launch_gemm_ntw<2>(s, epi, A, g_wpack, bias, add, out, M, N, K);
}

struct Scr {
  float *x0, *xbuf, *xf, *xb, *xl;
  float *x6[6];
  float *r, *wz, *k, *v, *asig, *g, *kk, *vs;
  float *wt, *at, *vt, *gt;
  float *vfirst, *wkvo, *od, *kb, *chid;
  float *h256, *h512, *hl, *u, *mm, *r1, *r2;
};

// Per-layer tensor offsets inside the 33-entry layer block (insertion order):
// 0 ln1_w 1 ln1_b 2 ln2_w 3 ln2_b | tmix: 4 x_r 5 x_w 6 x_k 7 x_v 8 x_a 9 x_g
// 10 w0 11 w1 12 w2 13 a0 14 a1 15 a2 16 v0 17 v1 18 v2 19 g1 20 g2
// 21 k_k 22 k_a 23 r_k 24 Wr 25 Wk 26 Wv 27 Wo 28 ln_w 29 ln_b
// cmix: 30 x_k 31 Wk 32 Wv
static void run_core(hipStream_t s, void* const* d_in, int base, float* x, Scr& W) {
  auto P = [&](int i) -> const float* { return (const float*)d_in[i]; };
  const int NE = NTOK * HD;
  const dim3 eb(256), eg((NE + 255) / 256);
  for (int l = 0; l < 6; ++l) {
    const int lb = base + l * 33;
    layernorm_k<<<NTOK, 32, 0, s>>>(x, P(lb + 0), P(lb + 1), W.xl, HD);
    mix6_k<<<NTOK, HD, 0, s>>>(W.xl, P(lb + 4), P(lb + 5), P(lb + 6), P(lb + 7),
                               P(lb + 8), P(lb + 9),
                               W.x6[0], W.x6[1], W.x6[2], W.x6[3], W.x6[4], W.x6[5]);
    // r = xr @ Wr
    launch_gemm(s, EPI_NONE, W.x6[0], P(lb + 24), nullptr, nullptr, W.r, NTOK, HD, HD);
    // w = -softplus(-(w0 + tanh(xw@w1)@w2)) - 0.5
    launch_gemm(s, EPI_TANH, W.x6[1], P(lb + 11), nullptr, nullptr, W.wt, NTOK, 32, HD);
    launch_gemm(s, EPI_BIAS, W.wt, P(lb + 12), P(lb + 10), nullptr, W.wz, NTOK, HD, 32);
    ew_wdecay<<<eg, eb, 0, s>>>(W.wz, NE);
    // k, v
    launch_gemm(s, EPI_NONE, W.x6[2], P(lb + 25), nullptr, nullptr, W.k, NTOK, HD, HD);
    launch_gemm(s, EPI_NONE, W.x6[3], P(lb + 26), nullptr, nullptr, W.v, NTOK, HD, HD);
    if (l == 0) {
      ew_copy<<<eg, eb, 0, s>>>(W.v, W.vfirst, NE);
    } else {
      launch_gemm(s, EPI_NONE, W.x6[3], P(lb + 17), nullptr, nullptr, W.vt, NTOK, 32, HD);
      launch_gemm(s, EPI_BIAS_SIGMOID, W.vt, P(lb + 18), P(lb + 16), nullptr, W.vs, NTOK, HD, 32);
      ew_vmix<<<eg, eb, 0, s>>>(W.v, W.vfirst, W.vs, NE);
    }
    // a = sigmoid(a0 + (xa@a1)@a2)
    launch_gemm(s, EPI_NONE, W.x6[4], P(lb + 14), nullptr, nullptr, W.at, NTOK, 32, HD);
    launch_gemm(s, EPI_BIAS_SIGMOID, W.at, P(lb + 15), P(lb + 13), nullptr, W.asig, NTOK, HD, 32);
    // g = sigmoid(xg@g1) @ g2
    launch_gemm(s, EPI_SIGMOID, W.x6[5], P(lb + 19), nullptr, nullptr, W.gt, NTOK, 64, HD);
    launch_gemm(s, EPI_NONE, W.gt, P(lb + 20), nullptr, nullptr, W.g, NTOK, HD, 64);
    // kk = normalize_head(k*k_k); k = k*(1+(a-1)*k_a)
    kk_k<<<NTOK * HNH, 32, 0, s>>>(W.k, W.asig, P(lb + 21), P(lb + 22), W.kk, W.k);
    // WKV recurrence
    wkv7_k<<<BQ * HNH, HEADSZ, 0, s>>>(W.r, W.wz, W.k, W.v, W.kk, W.asig, W.wkvo);
    // groupnorm + bonus + gate
    post_k<<<NTOK * HNH, 32, 0, s>>>(W.wkvo, P(lb + 28), P(lb + 29),
                                     W.r, W.k, W.v, W.g, P(lb + 23), W.od);
    // x = x + od @ Wo
    launch_gemm(s, EPI_ADD, W.od, P(lb + 27), nullptr, x, x, NTOK, HD, HD);
    // cmix
    layernorm_k<<<NTOK, 32, 0, s>>>(x, P(lb + 2), P(lb + 3), W.xl, HD);
    mix1_k<<<NTOK, HD, 0, s>>>(W.xl, P(lb + 30), W.kb);
    launch_gemm(s, EPI_RELU_SQ, W.kb, P(lb + 31), nullptr, nullptr, W.chid, NTOK, 4 * HD, HD);
    launch_gemm(s, EPI_ADD, W.chid, P(lb + 32), nullptr, x, x, NTOK, HD, 4 * HD);
  }
}

extern "C" void kernel_launch(void* const* d_in, const int* in_sizes, int n_in,
                              void* d_out, int out_size, void* d_ws, size_t ws_size,
                              hipStream_t stream) {
  (void)in_sizes; (void)n_in; (void)out_size; (void)ws_size;
  auto P = [&](int i) -> const float* { return (const float*)d_in[i]; };
  const float* zmix = P(0);

  // ---- workspace bump allocator (deterministic layout) ----
  float* wsf = (float*)d_ws;
  size_t off = 0;
  auto alloc = [&](size_t n) { float* p = wsf + off; off += (n + 7) & ~(size_t)7; return p; };
  Scr W;
  const size_t E = (size_t)NTOK * HD;
  W.x0 = alloc(E);  W.xbuf = alloc(E); W.xf = alloc(E); W.xb = alloc(E); W.xl = alloc(E);
  for (int i = 0; i < 6; ++i) W.x6[i] = alloc(E);
  W.r = alloc(E); W.wz = alloc(E); W.k = alloc(E); W.v = alloc(E);
  W.asig = alloc(E); W.g = alloc(E); W.kk = alloc(E); W.vs = alloc(E);
  W.wt = alloc((size_t)NTOK * 32); W.at = alloc((size_t)NTOK * 32);
  W.vt = alloc((size_t)NTOK * 32); W.gt = alloc((size_t)NTOK * 64);
  W.vfirst = alloc(E); W.wkvo = alloc(E); W.od = alloc(E); W.kb = alloc(E);
  W.chid = alloc((size_t)NTOK * 1024);
  W.h256 = alloc(E); W.h512 = alloc((size_t)NTOK * CIN); W.hl = alloc((size_t)NTOK * CIN);
  W.u = alloc(E); W.mm = alloc((size_t)NTOK * 1024);
  W.r1 = alloc((size_t)NTOK * CIN); W.r2 = alloc((size_t)NTOK * CIN);
  g_wpack = (__bf16*)alloc((size_t)CIN * 1024 / 2 + 16);  // bf16 pack buffer (1MB max weight)

  const int NE = NTOK * HD;
  const dim3 eb(256), eg((NE + 255) / 256);

  // x0 = z_mix @ down_w + down_b
  launch_gemm(stream, EPI_BIAS, zmix, P(1), P(2), nullptr, W.x0, NTOK, HD, CIN);

  // forward core
  ew_copy<<<eg, eb, 0, stream>>>(W.x0, W.xbuf, NE);
  run_core(stream, d_in, 5, W.xbuf, W);
  ew_copy<<<eg, eb, 0, stream>>>(W.xbuf, W.xf, NE);

  // backward core on time-flipped input, then flip back
  flipT_k<<<eg, eb, 0, stream>>>(W.x0, W.xbuf, HD);
  run_core(stream, d_in, 5 + 6 * 33, W.xbuf, W);
  flipT_k<<<eg, eb, 0, stream>>>(W.xbuf, W.xb, HD);

  // h = 0.5*(xf+xb); h512 = h @ up_w + up_b
  ew_avg<<<eg, eb, 0, stream>>>(W.xf, W.xb, W.h256, NE);
  launch_gemm(stream, EPI_BIAS, W.h256, P(3), P(4), nullptr, W.h512, NTOK, CIN, HD);

  const int A_IDX = 5 + 12 * 33;        // alpha = 401
  const int H1 = A_IDX + 1, H2 = H1 + 6, MB = H2 + 6;
  const int NU = NTOK * HD;
  const dim3 ug((NU + 255) / 256);

  // snake head 1
  layernorm_k<<<NTOK, 32, 0, stream>>>(W.h512, P(H1 + 0), P(H1 + 1), W.hl, CIN);
  launch_gemm(stream, EPI_BIAS, W.hl, P(H1 + 2), P(H1 + 3), nullptr, W.u, NTOK, HD, CIN);
  snake_k<<<ug, eb, 0, stream>>>(W.u, P(A_IDX), NU);
  launch_gemm(stream, EPI_BIAS, W.u, P(H1 + 4), P(H1 + 5), nullptr, W.r1, NTOK, CIN, HD);

  // snake head 2
  layernorm_k<<<NTOK, 32, 0, stream>>>(W.h512, P(H2 + 0), P(H2 + 1), W.hl, CIN);
  launch_gemm(stream, EPI_BIAS, W.hl, P(H2 + 2), P(H2 + 3), nullptr, W.u, NTOK, HD, CIN);
  snake_k<<<ug, eb, 0, stream>>>(W.u, P(A_IDX), NU);
  launch_gemm(stream, EPI_BIAS, W.u, P(H2 + 4), P(H2 + 5), nullptr, W.r2, NTOK, CIN, HD);

  // mask logits + softmax-pair + combine
  layernorm_k<<<NTOK, 32, 0, stream>>>(W.h512, P(MB + 0), P(MB + 1), W.hl, CIN);
  launch_gemm(stream, EPI_BIAS, W.hl, P(MB + 2), P(MB + 3), nullptr, W.mm, NTOK, 2 * CIN, CIN);
  final_k<<<NTOK, 256, 0, stream>>>(zmix, W.mm, W.r1, W.r2, (float*)d_out);
}
